// HMMTrajNet_52621939310683
// MI455X (gfx1250) — compile-verified, matrix operand order checked by reading, not run on hardware
//
#include <hip/hip_runtime.h>
#include <hip/hip_bf16.h>
#include <stdint.h>

#define B_SZ   256
#define MAXT   512
#define S_DIM  256
#define NCOL   832      // 640 action + 128 stop + 64 start
#define NT_T   52       // 832/16 column tiles
#define KT_T   8        // 256/32 K tiles
#define LSTR   833      // padded LDS row stride (odd -> no bank conflicts)
#define WB_ELEMS (NT_T*KT_T*32*16)   // 212992 bf16 elements
#define NCHUNK 33

typedef __attribute__((ext_vector_type(8)))  float          v8f;
typedef __attribute__((ext_vector_type(16))) __bf16         v16bf;
typedef __attribute__((ext_vector_type(16))) unsigned short v16u;

union V16 { v16u u; v16bf b; };

__device__ inline unsigned short f2bf(float x) {
    union { float f; unsigned u; } c; c.f = x;
    unsigned r = c.u + 0x7FFFu + ((c.u >> 16) & 1u);   // round-to-nearest-even
    return (unsigned short)(r >> 16);
}

__device__ inline float laddexp(float a, float b) {
    float m = fmaxf(a, b);
    return m + log1pf(__expf(-fabsf(a - b)));
}

// gfx1250 async DMA: 16B global -> LDS, tracked with ASYNCcnt
__device__ inline void async_g2l_b128(uint32_t ldsAddr, uint64_t gaddr) {
    asm volatile("global_load_async_to_lds_b128 %0, %1, off"
                 :: "v"(ldsAddr), "v"(gaddr) : "memory");
}

// ---------------------------------------------------------------------------
// Kernel 1: repack weights (f32, 'sbn'/'sbk'/'sb' layouts) into bf16 WMMA
// B-fragment layout for v_wmma_f32_16x16x32_bf16:
//   lane l: column n = l&15, khalf = l>>4
//   element e = 2v+p: K = kt*32 + (v>=4 ? 16 : 0) + (v&3)*2 + p + khalf*8
// Storage: wb[((nt*8 + kt)*32 + l)*16 + e]
// ---------------------------------------------------------------------------
__global__ void prep_wb(const float* __restrict__ Wa, const float* __restrict__ Ws,
                        const float* __restrict__ Wst, unsigned short* __restrict__ wb)
{
    int gid = blockIdx.x * 256 + threadIdx.x;
    if (gid >= WB_ELEMS) return;
    int nt  = gid >> 12;           // / 4096
    int rem = gid & 4095;
    int kt  = rem >> 9;
    int l   = (rem >> 4) & 31;
    int e   = rem & 15;
    int v = e >> 1, p = e & 1;
    int k   = kt * 32 + ((v >= 4) ? 16 : 0) + ((v & 3) * 2 + p) + (l >> 4) * 8;
    int col = nt * 16 + (l & 15);
    float val;
    if (col < 640)      val = Wa [k * 640 + col];          // (S,NB,NA) flat: k*640 + b*10+n
    else if (col < 768) val = Ws [k * 128 + (col - 640)];  // (S,NB,2)
    else                val = Wst[k * 64  + (col - 768)];  // (S,NB)
    wb[gid] = f2bf(val);
}

// ---------------------------------------------------------------------------
// Kernel 2: fused GEMM (bf16 WMMA) + log_softmax + gather + HMM forward scan.
// One workgroup per batch element; 4 waves; 33 chunks of 16 timesteps.
// A chunks are double-buffered in LDS via global_load_async_to_lds_b128.
// ---------------------------------------------------------------------------
__global__ void __launch_bounds__(128) hmm_fwd(
    const float* __restrict__ sI, const int* __restrict__ acts,
    const int* __restrict__ lens, const unsigned short* __restrict__ wbRaw,
    float* __restrict__ totals)
{
    extern __shared__ float smem[];
    float* sRaw   = smem;                  // 2 x 16x256 f32 (double-buffered A)
    float* lg     = sRaw + 2 * 16 * 256;   // 16*LSTR logits
    float* aG     = lg + 16 * LSTR;        // 16*64 gathered action logp
    float* sBeta  = aG + 16 * 64;          // 16*64 stop[STOP]
    float* sOmb   = sBeta + 16 * 64;       // 16*64 stop[CONTINUE]
    float* sStart = sOmb + 16 * 64;        // 16*64 start logp
    float* rowLse = sStart + 16 * 64;      // 16
    int*   sActs  = (int*)(rowLse + 16);   // 16

    const int b    = blockIdx.x;
    const int tid  = threadIdx.x;
    const int lane = tid & 31;
    const int wid  = tid >> 5;
    const int lm   = lane & 15;
    const int kh   = lane >> 4;
    const int len  = lens[b];
    const v16u* wb = (const v16u*)wbRaw;
    const int ntBase = wid * 13;

    float f0 = 0.f, f1 = 0.f, x00 = 0.f, x01 = 0.f, xb0 = 0.f, xb1 = 0.f;

    // ---- issue async DMA for chunk 0 into buffer 0 ----
    // 16 rows x 1KB each; 128 threads x 8 x 16B segments per chunk
    {
        const uint32_t ldsBase = (uint32_t)(uintptr_t)sRaw;
        #pragma unroll
        for (int i = 0; i < 8; ++i) {
            int seg = tid + i * 128;          // 0..1023
            int row = seg >> 6;               // 64 x 16B per row
            int off = (seg & 63) * 16;        // bytes within row
            int gr  = row; /* t0 == 0 */      // rows 0..15 all valid
            uint64_t g = (uint64_t)(uintptr_t)
                ((const char*)(sI + ((size_t)b * (MAXT + 1) + gr) * S_DIM) + off);
            async_g2l_b128(ldsBase + (uint32_t)(row * 1024 + off), g);
        }
    }

    for (int chunk = 0; chunk < NCHUNK; ++chunk) {
        const int t0  = chunk * 16;
        const int cur = chunk & 1;

        // ---- kick off async DMA for the *next* chunk, then wait for current
        if (chunk + 1 < NCHUNK) {
            const int nt0 = t0 + 16;
            const uint32_t ldsBase =
                (uint32_t)(uintptr_t)(sRaw + ((chunk + 1) & 1) * 16 * 256);
            #pragma unroll
            for (int i = 0; i < 8; ++i) {
                int seg = tid + i * 128;
                int row = seg >> 6;
                int off = (seg & 63) * 16;
                int gr  = nt0 + row; if (gr > MAXT) gr = MAXT;   // clamp tail
                uint64_t g = (uint64_t)(uintptr_t)
                    ((const char*)(sI + ((size_t)b * (MAXT + 1) + gr) * S_DIM) + off);
                async_g2l_b128(ldsBase + (uint32_t)(row * 1024 + off), g);
            }
            asm volatile("s_wait_asynccnt 0x8" ::: "memory"); // current chunk resident
        } else {
            asm volatile("s_wait_asynccnt 0x0" ::: "memory");
        }
        __syncthreads();   // make async-written LDS visible to all waves

        // ---- GEMM: A(16x256, LDS) x W(256x832, L2-resident bf16), f32 acc ----
        const float* arow = sRaw + cur * 16 * 256 + lm * S_DIM;

        v8f acc[13];
        #pragma unroll
        for (int i = 0; i < 13; ++i) acc[i] = (v8f){0.f,0.f,0.f,0.f,0.f,0.f,0.f,0.f};

        for (int kt = 0; kt < KT_T; ++kt) {
            const int k0 = kt * 32 + kh * 8;
            float4 fa = *(const float4*)(arow + k0);
            float4 fb = *(const float4*)(arow + k0 + 4);
            float4 fc = *(const float4*)(arow + k0 + 16);
            float4 fd = *(const float4*)(arow + k0 + 20);
            V16 A;
            A.u[0]=f2bf(fa.x);  A.u[1]=f2bf(fa.y);  A.u[2]=f2bf(fa.z);  A.u[3]=f2bf(fa.w);
            A.u[4]=f2bf(fb.x);  A.u[5]=f2bf(fb.y);  A.u[6]=f2bf(fb.z);  A.u[7]=f2bf(fb.w);
            A.u[8]=f2bf(fc.x);  A.u[9]=f2bf(fc.y);  A.u[10]=f2bf(fc.z); A.u[11]=f2bf(fc.w);
            A.u[12]=f2bf(fd.x); A.u[13]=f2bf(fd.y); A.u[14]=f2bf(fd.z); A.u[15]=f2bf(fd.w);

            if (kt < KT_T - 1)   // pull next K-step's B fragments toward L0/L2
                __builtin_prefetch((const void*)&wb[ntBase * 256 + (kt + 1) * 32 + lane], 0, 1);

            #pragma unroll
            for (int i = 0; i < 13; ++i) {
                V16 Bf; Bf.u = wb[(ntBase + i) * 256 + kt * 32 + lane];
                acc[i] = __builtin_amdgcn_wmma_f32_16x16x32_bf16(
                    false, A.b, false, Bf.b, (short)0, acc[i], false, false);
            }
        }

        // ---- D tiles -> LDS logits (lane holds col=nt*16+lm, rows kh*8+r) ----
        #pragma unroll
        for (int i = 0; i < 13; ++i) {
            const int col = (ntBase + i) * 16 + lm;
            #pragma unroll
            for (int r = 0; r < 8; ++r)
                lg[(kh * 8 + r) * LSTR + col] = acc[i][r];
        }
        if (tid < 16) {
            int t = t0 + tid;
            sActs[tid] = acts[b * MAXT + (t < MAXT ? t : MAXT - 1)];
        }
        __syncthreads();

        // ---- per-row logsumexp over 64 start logits ----
        if (tid < 16) {
            const float* rp = lg + tid * LSTR + 768;
            float m = -INFINITY;
            for (int j = 0; j < 64; ++j) m = fmaxf(m, rp[j]);
            float s = 0.f;
            for (int j = 0; j < 64; ++j) s += __expf(rp[j] - m);
            rowLse[tid] = m + __logf(s);
        }
        __syncthreads();

        // ---- softmaxes + action gather: 16 rows x 64 options ----
        for (int idx = tid; idx < 1024; idx += 128) {
            const int r = idx >> 6, bb = idx & 63;
            const float* rp = lg + r * LSTR;
            {   // action log_softmax over NA=10, gather taken action
                const float* ap = rp + bb * 10;
                float m = ap[0];
                #pragma unroll
                for (int n = 1; n < 10; ++n) m = fmaxf(m, ap[n]);
                float s = 0.f;
                #pragma unroll
                for (int n = 0; n < 10; ++n) s += __expf(ap[n] - m);
                aG[r * 64 + bb] = ap[sActs[r]] - m - __logf(s);
            }
            {   // stop log_softmax over 2
                float xs = rp[640 + 2 * bb], xc = rp[640 + 2 * bb + 1];
                float l2 = laddexp(xs, xc);
                sBeta[r * 64 + bb] = xs - l2;
                sOmb [r * 64 + bb] = xc - l2;
            }
            sStart[r * 64 + bb] = rp[768 + bb] - rowLse[r];
        }
        __syncthreads();

        // ---- HMM forward scan: wave 0, 2 states per lane, shfl butterfly ----
        if (tid < 32) {
            const int j0 = lane, j1 = lane + 32;
            for (int r = 0; r < 16; ++r) {
                const int t = t0 + r;
                if (t > MAXT) break;
                if (t == len) { xb0 = sBeta[r * 64 + j0]; xb1 = sBeta[r * 64 + j1]; }
                if (t == 0) {
                    f0 = sStart[j0] + aG[j0];
                    f1 = sStart[j1] + aG[j1];
                } else if (t < MAXT) {
                    // M = logsumexp_i(f_prev[i] + beta_t[i])  (rank-1 term)
                    float v0 = f0 + sBeta[r * 64 + j0];
                    float v1 = f1 + sBeta[r * 64 + j1];
                    float m = fmaxf(v0, v1);
                    float s = __expf(v0 - m) + __expf(v1 - m);
                    #pragma unroll
                    for (int off = 16; off > 0; off >>= 1) {
                        float om = __shfl_xor(m, off, 32);
                        float os = __shfl_xor(s, off, 32);
                        float nm = fmaxf(m, om);
                        s = s * __expf(m - nm) + os * __expf(om - nm);
                        m = nm;
                    }
                    float M = m + __logf(s);
                    f0 = laddexp(M + sStart[r*64+j0], f0 + sOmb[r*64+j0]) + aG[r*64+j0];
                    f1 = laddexp(M + sStart[r*64+j1], f1 + sOmb[r*64+j1]) + aG[r*64+j1];
                }
                if (t == len - 1) { x00 = f0; x01 = f1; }
            }
        }
        __syncthreads();
    }

    // ---- total_logps[b] = logsumexp_j(f[len-1][j] + beta[len][j]) ----
    if (tid < 32) {
        float v0 = x00 + xb0, v1 = x01 + xb1;
        float m = fmaxf(v0, v1);
        float s = __expf(v0 - m) + __expf(v1 - m);
        #pragma unroll
        for (int off = 16; off > 0; off >>= 1) {
            float om = __shfl_xor(m, off, 32);
            float os = __shfl_xor(s, off, 32);
            float nm = fmaxf(m, om);
            s = s * __expf(m - nm) + os * __expf(om - nm);
            m = nm;
        }
        if (lane == 0) totals[b] = m + __logf(s);
    }
}

// ---------------------------------------------------------------------------
// Kernel 3: deterministic reduction, out = -sum(totals)
// ---------------------------------------------------------------------------
__global__ void finalize_sum(const float* __restrict__ totals, float* __restrict__ out)
{
    __shared__ float red[256];
    int t = threadIdx.x;
    red[t] = totals[t];
    __syncthreads();
    for (int off = 128; off > 0; off >>= 1) {
        if (t < off) red[t] += red[t + off];
        __syncthreads();
    }
    if (t == 0) out[0] = -red[0];
}

extern "C" void kernel_launch(void* const* d_in, const int* in_sizes, int n_in,
                              void* d_out, int out_size, void* d_ws, size_t ws_size,
                              hipStream_t stream)
{
    const float* sI   = (const float*)d_in[0];   // (B, T+1, S) f32
    const int*   acts = (const int*)d_in[1];     // (B, T) i32
    const int*   lens = (const int*)d_in[2];     // (B,) i32
    const float* Wa   = (const float*)d_in[3];   // (S, NB, NA)
    const float* Ws   = (const float*)d_in[4];   // (S, NB, 2)
    const float* Wst  = (const float*)d_in[5];   // (S, NB)

    unsigned short* wb = (unsigned short*)d_ws;                      // 425984 B
    float* totals = (float*)((char*)d_ws + (size_t)WB_ELEMS * 2);    // 256 f32

    prep_wb<<<(WB_ELEMS + 255) / 256, 256, 0, stream>>>(Wa, Ws, Wst, wb);

    size_t smemBytes = (size_t)(2 * 16 * 256 + 16 * LSTR + 4 * 16 * 64 + 16) * sizeof(float)
                     + 16 * sizeof(int);
    (void)hipFuncSetAttribute((const void*)hmm_fwd,
                              hipFuncAttributeMaxDynamicSharedMemorySize,
                              (int)smemBytes);
    hmm_fwd<<<B_SZ, 128, smemBytes, stream>>>(sI, acts, lens, wb, totals);

    finalize_sum<<<1, 256, 0, stream>>>(totals, (float*)d_out);
}
